// CalibratedProjectiveLinear_14293651161455
// MI455X (gfx1250) — matrix operand order repaired and checked by reading, not compiled, not run on hardware
//
#include <hip/hip_runtime.h>
#include <hip/hip_bf16.h>

typedef __attribute__((ext_vector_type(16))) _Float16 v16h;
typedef __attribute__((ext_vector_type(8)))  _Float16 v8h;
typedef __attribute__((ext_vector_type(4)))  _Float16 v4h;
typedef __attribute__((ext_vector_type(8)))  float    v8f;

union FragU { v16h v16; v8h v8[2]; };

constexpr int BSZ = 512;     // batch rows
constexpr int IN  = 4096;    // reduction dim
constexpr int OUT = 11008;   // output channels

constexpr int TM  = 128;     // x-rows per block
constexpr int TN  = 128;     // out-channels per block
constexpr int TK  = 32;      // K slice per iteration (WMMA f16 K)
constexpr int LDK = 40;      // padded LDS row length in halfs (80B, 16B-aligned rows)

// Exact power-of-2 rescale keeps all f16 operands normal:
// W in [e^-10, 1]: LUT stores W*1024 (>= 0.046), x converted as x/1024.
// Products and the f32 WMMA accumulation are unchanged.
constexpr float WSCALE = 1024.0f;
constexpr float XSCALE = 1.0f / 1024.0f;

__global__ __launch_bounds__(256)
void qins_wmma_gemm(const float* __restrict__ x,
                    const int*   __restrict__ stored,
                    const int*   __restrict__ sign,
                    const float* __restrict__ log_min_p,
                    const float* __restrict__ log_max_p,
                    const float* __restrict__ scale,
                    const float* __restrict__ bias,
                    float*       __restrict__ out)
{
    __shared__ __align__(16) _Float16 As[TM][LDK];   // x tile   (m-major, K contiguous)
    __shared__ __align__(16) _Float16 Bs[TN][LDK];   // W tile   (o-major, K contiguous)
    __shared__ float lut[256];                       // exp LUT * WSCALE (stable after fill)

    const int tid    = threadIdx.x;       // 0..255
    const int lane   = tid & 31;          // wave32
    const int wave   = tid >> 5;          // 0..7
    const int wave_m = wave & 3;          // 4 wave-rows * 32 rows
    const int wave_n = wave >> 2;         // 2 wave-cols * 64 cols

    const int ocol0 = blockIdx.x * TN;    // OUT/TN = 86, exact
    const int brow0 = blockIdx.y * TM;    // BSZ/TM = 4,  exact

    // 256-entry dequant LUT (replaces 45M v_exp_f32 with ds gathers).
    {
        const float lmin = log_min_p[0];
        const float lmax = log_max_p[0];
        const float norm = (255.0f - (float)tid) * (1.0f / 254.0f);
        lut[tid] = expf(lmin + norm * (lmax - lmin)) * WSCALE;
    }

    v8f acc[2][4];
    #pragma unroll
    for (int i = 0; i < 2; ++i)
        #pragma unroll
        for (int j = 0; j < 4; ++j)
            acc[i][j] = v8f{};

    // Cooperative load mapping: 256 threads cover a 32-row x 32-col slab per pass.
    const int lrow = tid >> 3;          // 0..31
    const int lcol = (tid & 7) * 4;     // 0,4,...,28 (float4 / int4 granules)

    // Fragment lane decomposition (gfx1250 wave32 WMMA layouts).
    const int lm = lane & 15;
    const int kA = (lane < 16) ? 0 : 8;   // A: VGPR0-3 -> K kA..kA+7, VGPR4-7 -> 16+kA..
    const int kB = (lane < 16) ? 0 : 8;   // B: VGPR0-3 -> K kB..kB+7, VGPR4-7 -> 16+kB..

    // Software-pipeline registers: raw next-tile data in flight during WMMA.
    float4 xr[4];
    int4   sr[4];
    int4   gr[4];

    // Prologue: issue loads for kt = 0.
    #pragma unroll
    for (int i = 0; i < 4; ++i) {
        const int m = lrow + 32 * i;
        xr[i] = *(const float4*)&x[(size_t)(brow0 + m) * IN + lcol];
        const size_t wb = (size_t)(ocol0 + m) * IN + lcol;
        sr[i] = *(const int4*)&stored[wb];
        gr[i] = *(const int4*)&sign[wb];
    }

    __syncthreads();   // LUT visible to all waves before first dequant

    for (int kt = 0; kt < IN; kt += TK) {
        // ---- store phase: dequant in-register tile -> LDS -------------------
        #pragma unroll
        for (int i = 0; i < 4; ++i) {
            const int m = lrow + 32 * i;
            {
                v4h h;
                h[0] = (_Float16)(xr[i].x * XSCALE);
                h[1] = (_Float16)(xr[i].y * XSCALE);
                h[2] = (_Float16)(xr[i].z * XSCALE);
                h[3] = (_Float16)(xr[i].w * XSCALE);
                *(v4h*)&As[m][lcol] = h;
            }
            {
                v4h h;
                h[0] = (_Float16)((float)gr[i].x * lut[sr[i].x]);
                h[1] = (_Float16)((float)gr[i].y * lut[sr[i].y]);
                h[2] = (_Float16)((float)gr[i].z * lut[sr[i].z]);
                h[3] = (_Float16)((float)gr[i].w * lut[sr[i].w]);
                *(v4h*)&Bs[m][lcol] = h;
            }
        }

        __syncthreads();   // tiles visible before fragment reads

        // ---- prefetch next K-tile: loads fly during the WMMA phase ---------
        const int kn = kt + TK;
        if (kn < IN) {
            #pragma unroll
            for (int i = 0; i < 4; ++i) {
                const int m = lrow + 32 * i;
                xr[i] = *(const float4*)&x[(size_t)(brow0 + m) * IN + kn + lcol];
                const size_t wb = (size_t)(ocol0 + m) * IN + kn + lcol;
                sr[i] = *(const int4*)&stored[wb];
                gr[i] = *(const int4*)&sign[wb];
            }
        }

        // ---- fragment loads: two aligned 16B LDS reads per fragment --------
        v16h af[2];
        #pragma unroll
        for (int im = 0; im < 2; ++im) {
            const int m = wave_m * 32 + im * 16 + lm;
            FragU u;
            u.v8[0] = *(const v8h*)&As[m][kA];        // a.h[0..7]  = A[m][kA..kA+7]
            u.v8[1] = *(const v8h*)&As[m][kA + 16];   // a.h[8..15] = A[m][16+kA..]
            af[im] = u.v16;
        }
        v16h bf[4];
        #pragma unroll
        for (int jn = 0; jn < 4; ++jn) {
            const int o = wave_n * 64 + jn * 16 + lm;
            FragU u;
            u.v8[0] = *(const v8h*)&Bs[o][kB];        // b.h[0..7]  = W[o][kB..kB+7]
            u.v8[1] = *(const v8h*)&Bs[o][kB + 16];   // b.h[8..15] = W[o][16+kB..]
            bf[jn] = u.v16;
        }

        // ---- 8 WMMAs per wave per K-step -----------------------------------
        #pragma unroll
        for (int im = 0; im < 2; ++im)
            #pragma unroll
            for (int jn = 0; jn < 4; ++jn)
                acc[im][jn] = __builtin_amdgcn_wmma_f32_16x16x32_f16(
                    false, af[im], false, bf[jn],
                    (short)0, acc[im][jn], false, false);

        __syncthreads();   // all waves done reading LDS before next overwrite
    }

    // ---- epilogue: (acc + bias[o]) * scale[o] -------------------------------
    // C/D layout: lane holds column N = lane&15; VGPR r = row M (lanes 16-31: M+8).
    #pragma unroll
    for (int jn = 0; jn < 4; ++jn) {
        const int o  = ocol0 + wave_n * 64 + jn * 16 + (lane & 15);
        const float sc = scale[o];
        const float bi = bias[o];
        #pragma unroll
        for (int im = 0; im < 2; ++im) {
            const int mbase = brow0 + wave_m * 32 + im * 16 + ((lane < 16) ? 0 : 8);
            #pragma unroll
            for (int r = 0; r < 8; ++r)
                out[(size_t)(mbase + r) * OUT + o] = (acc[im][jn][r] + bi) * sc;
        }
    }
}

extern "C" void kernel_launch(void* const* d_in, const int* in_sizes, int n_in,
                              void* d_out, int out_size, void* d_ws, size_t ws_size,
                              hipStream_t stream) {
    // setup_inputs() order: x, stored, sign, log_min, log_max, scale, bias
    const float* x       = (const float*)d_in[0];
    const int*   stored  = (const int*)  d_in[1];
    const int*   sign    = (const int*)  d_in[2];
    const float* log_min = (const float*)d_in[3];
    const float* log_max = (const float*)d_in[4];
    const float* scale   = (const float*)d_in[5];
    const float* bias    = (const float*)d_in[6];
    float* out = (float*)d_out;

    dim3 grid(OUT / TN, BSZ / TM);   // (86, 4) blocks, exact tiling
    qins_wmma_gemm<<<grid, 256, 0, stream>>>(x, stored, sign, log_min, log_max,
                                             scale, bias, out);
}
